// ARD_82360292868538
// MI455X (gfx1250) — matrix operand order repaired and checked by reading, not compiled, not run on hardware
//
#include <hip/hip_runtime.h>

typedef __attribute__((ext_vector_type(16))) __bf16        v16bf;
typedef __attribute__((ext_vector_type(8)))  float         v8f;
typedef __attribute__((ext_vector_type(4)))  unsigned int  v4u;

#define ARD_D 256          // feature dim, fixed by the reference
#define KSTEPS (ARD_D / 32)
#define B_ROW_PITCH 80     // padded LDS pitch (64B row + 16B pad -> conflict spread)
#define B_BUF_BYTES (64 * B_ROW_PITCH)   // 5120 B per buffer

// ---------------------------------------------------------------------------
// Kernel 1: scale rows by exp(-0.5*log_bw), emit bf16 rows + f32 squared norms
// ---------------------------------------------------------------------------
__global__ __launch_bounds__(256) void ard_scale_rows(
    const float* __restrict__ src,       // [rows, D] f32
    const float* __restrict__ lbw,       // [D] f32
    __bf16*      __restrict__ dst,       // [rows, D] bf16
    float*       __restrict__ sumsq)     // [rows] f32
{
    __shared__ float red[8];
    const int row = blockIdx.x;
    const int d   = threadIdx.x;                       // D == blockDim.x == 256
    const float w = __expf(-0.5f * lbw[d]);
    const float v = src[(size_t)row * ARD_D + d] * w;
    dst[(size_t)row * ARD_D + d] = (__bf16)v;

    float s = v * v;
    #pragma unroll
    for (int off = 16; off > 0; off >>= 1)
        s += __shfl_down(s, off, 32);
    if ((threadIdx.x & 31) == 0) red[threadIdx.x >> 5] = s;
    __syncthreads();
    if (threadIdx.x == 0) {
        float t = 0.0f;
        #pragma unroll
        for (int i = 0; i < 8; ++i) t += red[i];
        sumsq[row] = t;
    }
}

// ---------------------------------------------------------------------------
// Kernel 2: WMMA bf16 GEMM fused with ARD epilogue.
// 256 threads = 8 waves; WG tile 128x64 (wave: 16 rows x 64 cols).
// B tile (64x32 bf16): tile0 staged with visible ds_store_b128, tiles 1..7
// streamed with async global->LDS loads, double-buffered.
// ---------------------------------------------------------------------------
__global__ __launch_bounds__(256) void ard_wmma_gemm(
    const __bf16* __restrict__ xw,   // [N, D] bf16
    const __bf16* __restrict__ yw,   // [M, D] bf16
    const float*  __restrict__ x2,   // [N]
    const float*  __restrict__ y2,   // [M]
    float*        __restrict__ out,  // [N, M]
    int Mcols)
{
    __shared__ __align__(16) char bsm[2 * B_BUF_BYTES];

    const int tid  = threadIdx.x;
    const int lane = tid & 31;
    const int wave = tid >> 5;
    const int i0   = blockIdx.y * 128 + wave * 16;   // wave's output row base
    const int j0   = blockIdx.x * 64;                // WG output col base
    const int nrow = lane & 15;                      // row/col index within 16
    const int half = lane >> 4;                      // K-half selector

    // Cooperative staging map: thread t owns 16B chunk bp of B-row bj.
    const int bj = tid >> 2;                          // 0..63
    const int bp = tid & 3;                           // 0..3
    const unsigned ldsRowOff = (unsigned)(bj * B_ROW_PITCH + bp * 16);
    const __bf16* bGlobRow = yw + (size_t)(j0 + bj) * ARD_D + (bp << 3);

    // Prologue: tile 0 via compiler-visible loads + ds stores (guarantees bsm
    // has real writers so its ds_loads cannot be undef-folded).
    *(v4u*)(bsm + ldsRowOff) = *(const v4u*)bGlobRow;

    // Tile 1 -> buffer 1 via async global->LDS (bsm address escapes into asm).
    {
        unsigned ldsoff = (unsigned)B_BUF_BYTES + ldsRowOff;
        const __bf16* g = bGlobRow + 32;
        asm volatile("global_load_async_to_lds_b128 %0, %1, off"
                     :: "v"(ldsoff), "v"(g), "v"((const void*)bsm) : "memory");
    }

    // A fragment for kt=0 (ISA 7.12.2: two contiguous b128 per lane).
    const __bf16* aRow = xw + (size_t)(i0 + nrow) * ARD_D + (half << 3);
    union { v16bf v; v4u q[2]; } acur, anext;
    acur.q[0] = *(const v4u*)(aRow);
    acur.q[1] = *(const v4u*)(aRow + 16);

    __syncthreads();   // tile 0 visible to all waves

    v8f acc[4] = {};

    #pragma unroll
    for (int kt = 0; kt < KSTEPS; ++kt) {
        // Prefetch next A fragment into registers.
        if (kt + 1 < KSTEPS) {
            anext.q[0] = *(const v4u*)(aRow + (kt + 1) * 32);
            anext.q[1] = *(const v4u*)(aRow + (kt + 1) * 32 + 16);
        }

        // Hoist all 4 B fragments from buffer kt&1 (one ds clause, staged
        // dscnt waits can overlap the WMMAs below).
        const char* bbuf = bsm + (kt & 1) * B_BUF_BYTES;
        union { v16bf v; v4u q[2]; } bfrag[4];
        #pragma unroll
        for (int nb = 0; nb < 4; ++nb) {
            const char* brow = bbuf + (nb * 16 + nrow) * B_ROW_PITCH + half * 16;
            bfrag[nb].q[0] = *(const v4u*)(brow);
            bfrag[nb].q[1] = *(const v4u*)(brow + 32);
        }

        #pragma unroll
        for (int nb = 0; nb < 4; ++nb) {
            acc[nb] = __builtin_amdgcn_wmma_f32_16x16x32_bf16(
                false, acur.v, false, bfrag[nb].v, (short)0, acc[nb], false, false);
        }

        // Tile kt+1 resident (asynccnt) and buffer kt&1 drained by all waves
        // (barrier) -> safe to stream tile kt+2 into buffer kt&1.
        asm volatile("s_wait_asynccnt 0x0" ::: "memory");
        __syncthreads();
        if (kt + 2 < KSTEPS) {
            unsigned ldsoff = (unsigned)((kt & 1) * B_BUF_BYTES) + ldsRowOff;
            const __bf16* g = bGlobRow + (kt + 2) * 32;
            asm volatile("global_load_async_to_lds_b128 %0, %1, off"
                         :: "v"(ldsoff), "v"(g), "v"((const void*)bsm) : "memory");
        }
        acur = anext;
    }

    // Epilogue. C/D layout: VGPR r, lane -> row m = r + 8*half, col n = lane&15.
    float xi[8];
    #pragma unroll
    for (int r = 0; r < 8; ++r) xi[r] = x2[i0 + r + (half << 3)];

    #pragma unroll
    for (int nb = 0; nb < 4; ++nb) {
        const int j  = j0 + nb * 16 + nrow;
        const float yj = y2[j];
        #pragma unroll
        for (int r = 0; r < 8; ++r) {
            const int m = i0 + r + (half << 3);
            float p = xi[r] + yj - 2.0f * acc[nb][r];
            p = fmaxf(p, 0.0f);
            out[(size_t)m * Mcols + j] = __expf(-0.5f * p);
        }
    }
}

// ---------------------------------------------------------------------------
// Launch
// ---------------------------------------------------------------------------
extern "C" void kernel_launch(void* const* d_in, const int* in_sizes, int n_in,
                              void* d_out, int out_size, void* d_ws, size_t ws_size,
                              hipStream_t stream) {
    const float* x   = (const float*)d_in[0];
    const float* y   = (const float*)d_in[1];
    const float* lbw = (const float*)d_in[2];
    float*       out = (float*)d_out;

    const int D = in_sizes[2];            // 256
    const int N = in_sizes[0] / D;        // 4096
    const int M = in_sizes[1] / D;        // 4096

    // Workspace: xw (bf16 NxD) | yw (bf16 MxD) | x2 (f32 N) | y2 (f32 M)
    char*   ws = (char*)d_ws;
    __bf16* xw = (__bf16*)ws;
    __bf16* yw = (__bf16*)(ws + (size_t)N * D * sizeof(__bf16));
    float*  x2 = (float*)(ws + (size_t)(N + M) * D * sizeof(__bf16));
    float*  y2 = x2 + N;

    ard_scale_rows<<<N, 256, 0, stream>>>(x, lbw, xw, x2);
    ard_scale_rows<<<M, 256, 0, stream>>>(y, lbw, yw, y2);

    dim3 grid(M / 64, N / 128);           // 128x64 output tile per WG
    ard_wmma_gemm<<<grid, 256, 0, stream>>>(xw, yw, x2, y2, out, M);
}
